// RoPEMultiHeadAttention_7258494730299
// MI455X (gfx1250) — matrix operand order repaired and checked by reading, not compiled, not run on hardware
//
#include <hip/hip_runtime.h>
#include <stdint.h>

typedef __attribute__((ext_vector_type(16))) _Float16 v16h;
typedef __attribute__((ext_vector_type(8)))  float    v8f;
typedef __attribute__((ext_vector_type(4)))  unsigned int v4u;
typedef __attribute__((ext_vector_type(8)))  int      v8i;
typedef __attribute__((ext_vector_type(4)))  int      v4i;

#define B_  2
#define S_  2048
#define D_  1024
#define H_  16
#define HD_ 64

#define BST_LD 40    // padded row stride (halfs) for transposed weight tiles: 80B = 5x16B
#define VT_LD  272   // padded row stride (halfs) for TDM-staged V^T chunk: 544B = 34x16B

#define WMMA_F16(a, b, c) \
  __builtin_amdgcn_wmma_f32_16x16x32_f16(false, (a), false, (b), (short)0, (c), false, false)

union AFrag { uint4 q[2]; v16h h; };

// ---------------------------------------------------------------------------
// WMMA fragment gathers (all contiguous -> 2x 16B loads each).
// A (16x32 f16): lane l, row m=l&15; k = {hf*8+0..7, 16+hf*8+0..7} (hf=l>>4).
// B via transposed source M[n][k] (logical B[k][n]): lane n=l&15 reads
//   k = {hf*8+0..7, 16+hf*8+0..7} in row n (same order as A).
// C/D (16x16 f32): VGPR r, lane l: m=(l>>4)*8+r, n=l&15.
// ---------------------------------------------------------------------------
__device__ __forceinline__ v16h load_a_f16(const _Float16* A, int lda, int lane) {
  const int m = lane & 15, hf = lane >> 4;
  AFrag f;
  f.q[0] = *(const uint4*)(A + (size_t)m * lda + (hf << 3));
  f.q[1] = *(const uint4*)(A + (size_t)m * lda + 16 + (hf << 3));
  return f.h;
}

__device__ __forceinline__ v16h load_bT_f16(const _Float16* M, int ldm, int lane) {
  const int n = lane & 15, hf = lane >> 4;
  AFrag f;
  f.q[0] = *(const uint4*)(M + (size_t)n * ldm + (hf << 3));
  f.q[1] = *(const uint4*)(M + (size_t)n * ldm + 16 + (hf << 3));
  return f.h;
}

// ---------------------------------------------------------------------------
// Cooperative tile staging (block = 128 threads).
// A tile 16x32 row-major; B tile 32x64 stored TRANSPOSED as bsT[64][BST_LD].
// ---------------------------------------------------------------------------
__device__ __forceinline__ void stage_a_f32(_Float16* as, const float* A, int lda,
                                            int k0, int t) {
  const int idx = t * 4;
  const int r = idx >> 5, c = idx & 31;
  const float4 v = *(const float4*)(A + (size_t)r * lda + k0 + c);
  as[r * 32 + c + 0] = (_Float16)v.x;
  as[r * 32 + c + 1] = (_Float16)v.y;
  as[r * 32 + c + 2] = (_Float16)v.z;
  as[r * 32 + c + 3] = (_Float16)v.w;
}

__device__ __forceinline__ void stage_a_f16(_Float16* as, const _Float16* A, int lda,
                                            int k0, int t) {
  const int idx = t * 4;
  const int r = idx >> 5, c = idx & 31;
  *(uint2*)(as + r * 32 + c) = *(const uint2*)(A + (size_t)r * lda + k0 + c);
}

__device__ __forceinline__ void stage_bT_f32(_Float16* bsT, const float* Bm, int ldb,
                                             int k0, int t) {
#pragma unroll
  for (int u = 0; u < 4; ++u) {
    const int idx = (t + u * 128) * 4;
    const int r = idx >> 6, c = idx & 63;    // r = k row (0..31), c = n col (0..63)
    const float4 v = *(const float4*)(Bm + (size_t)(k0 + r) * ldb + c);
    bsT[(c + 0) * BST_LD + r] = (_Float16)v.x;
    bsT[(c + 1) * BST_LD + r] = (_Float16)v.y;
    bsT[(c + 2) * BST_LD + r] = (_Float16)v.z;
    bsT[(c + 3) * BST_LD + r] = (_Float16)v.w;
  }
}

// ---------------------------------------------------------------------------
// Tensor Data Mover: 2-D f16 tile copy global -> LDS (6-arg builtin form).
// D# group0: [1:0]=count=1, [63:32]=lds byte addr, [120:64]=global addr,
//            [127:126]=type=2.
// D# group1: data_size @[17:16]; pad_enable@20; pad_interval@[24:22];
//            pad_amount@[31:25]; tensor_dim0@[79:48]; tensor_dim1@[111:80];
//            tile_dim0@[127:112]; tile_dim1@[143:128]; stride0@[207:160].
// ---------------------------------------------------------------------------
__device__ __forceinline__ void tdm_load_2d_f16(const void* gptr, unsigned int lds_off,
                                                unsigned int tile_w, unsigned int tile_h,
                                                unsigned int ten_w, unsigned int ten_h,
                                                unsigned int stride0,
                                                unsigned int pad_en,
                                                unsigned int pad_interval_code,
                                                unsigned int pad_amount_code) {
  const uint64_t ga = (uint64_t)(uintptr_t)gptr;
  v4u g0;
  g0[0] = 1u;
  g0[1] = lds_off;
  g0[2] = (unsigned int)ga;
  g0[3] = (unsigned int)((ga >> 32) & 0x01ffffffu) | (2u << 30);
  v8i g1;
  g1[0] = (int)((1u << 16) | (pad_en << 20) | (pad_interval_code << 22)
                | (pad_amount_code << 25));
  g1[1] = (int)((ten_w & 0xffffu) << 16);
  g1[2] = (int)(((ten_w >> 16) & 0xffffu) | ((ten_h & 0xffffu) << 16));
  g1[3] = (int)(((ten_h >> 16) & 0xffffu) | ((tile_w & 0xffffu) << 16));
  g1[4] = (int)(tile_h & 0xffffu);
  g1[5] = (int)stride0;
  g1[6] = 0;
  g1[7] = 0;
  const v4i z4 = {0, 0, 0, 0};
  const v8i z8 = {0, 0, 0, 0, 0, 0, 0, 0};
  __builtin_amdgcn_tensor_load_to_lds(g0, g1, z4, z4, z8, 0);
}

__device__ __forceinline__ void tdm_load_1d_f16(const void* gptr, unsigned int lds_off,
                                                unsigned int nelem) {
  tdm_load_2d_f16(gptr, lds_off, nelem, 1, nelem, 1, nelem, 0, 0, 0);
}

#define LDSOFF(p) ((unsigned int)(uintptr_t)(p))

// ---------------------------------------------------------------------------
// Kernel 1: Q/K/V projection + bias + RoPE (Q,K). Block 128 thr, 16x64 tile.
// Q,K stored (B,H,S,Hd); V stored TRANSPOSED (B,H,Hd,S) for phase-3 TDM tiles.
// grid = (16 heads, 256 row-tiles, 3 matrices)
// ---------------------------------------------------------------------------
__global__ void qkv_rope_kernel(const float* __restrict__ X,
                                const float* __restrict__ Wq, const float* __restrict__ bq,
                                const float* __restrict__ Wk, const float* __restrict__ bk,
                                const float* __restrict__ Wv, const float* __restrict__ bv,
                                _Float16* __restrict__ qws, _Float16* __restrict__ kws,
                                _Float16* __restrict__ vtws) {
  __shared__ __align__(16) _Float16 as[16 * 32];
  __shared__ __align__(16) _Float16 bsT[64 * BST_LD];
  __shared__ float tile[16][64];

  const int t = threadIdx.x;
  const int lane = t & 31, wave = t >> 5;
  const int row0 = blockIdx.y * 16;
  const int head = blockIdx.x;
  const int col0 = head * 64 + wave * 16;
  const int z = blockIdx.z;
  const float* W    = (z == 0) ? Wq : (z == 1) ? Wk : Wv;
  const float* bias = (z == 0) ? bq : (z == 1) ? bk : bv;

  const float* Abase = X + (size_t)row0 * D_;
  const float* Bbase = W + head * 64;

  v8f acc = {};
  for (int k0 = 0; k0 < D_; k0 += 32) {
    stage_a_f32(as, Abase, D_, k0, t);
    stage_bT_f32(bsT, Bbase, D_, k0, t);
    __syncthreads();
    v16h a  = load_a_f16(as, 32, lane);
    v16h bf = load_bT_f16(bsT + (size_t)(wave * 16) * BST_LD, BST_LD, lane);
    acc = WMMA_F16(a, bf, acc);
    __syncthreads();
  }

  const float bval = bias[col0 + (lane & 15)];
  const int nloc = wave * 16 + (lane & 15);
#pragma unroll
  for (int r = 0; r < 8; ++r) {
    const int m = ((lane >> 4) << 3) + r;
    tile[m][nloc] = acc[r] + bval;
  }
  __syncthreads();

  for (int e = t; e < 16 * 64; e += 128) {
    // V uses (m = e&15, d = e>>4) so consecutive lanes write consecutive s.
    const int m = (z == 2) ? (e & 15) : (e >> 6);
    const int d = (z == 2) ? (e >> 4) : (e & 63);
    const int row = row0 + m;
    const int bidx = row >> 11;
    const int s = row & (S_ - 1);
    const float x = tile[m][d];
    if (z == 2) {
      vtws[((size_t)(bidx * H_ + head) * HD_ + d) * S_ + s] = (_Float16)x;
    } else {
      const int j = d & 31;
      const float inv = __expf(-(float)(2 * j) * (9.2103403719761836f / 64.0f));
      const float ang = (float)s * inv;
      const float c  = __cosf(ang);
      const float sn = __sinf(ang);
      const float partner = (d < 32) ? -tile[m][d + 32] : tile[m][d - 32];
      const float val = x * c + partner * sn;
      _Float16* dst = (z == 0) ? qws : kws;
      dst[((size_t)(bidx * H_ + head) * S_ + s) * HD_ + d] = (_Float16)val;
    }
  }
}

// ---------------------------------------------------------------------------
// Kernel 2 (fused): scores -> softmax -> attn write -> P@V -> ctx.
// Block 256 thr = 8 waves; one (bh, 16-row q-strip) per block.
// Phase 1: K streamed in 8x 256-key chunks (1-D TDM); 16 v8f score accs/wave.
// Phase 2: softmax via shfl_xor (16 lanes) + LDS (8 waves); probs written once.
// Phase 3: P chunks -> LDS f16; V^T chunks streamed as 2-D TDM tiles with LDS
//          row padding (VT_LD stride) -> contiguous b128 B-fragments;
//          4 col-tiles x 2-way split-K, partials combined via LDS.
// ---------------------------------------------------------------------------
__global__ void __launch_bounds__(256) attn_fused_kernel(
    const _Float16* __restrict__ qws, const _Float16* __restrict__ kws,
    const _Float16* __restrict__ vtws, float* __restrict__ attn,
    _Float16* __restrict__ ctxws) {
  __shared__ __align__(16) _Float16 qs[16 * 64];        // 2 KB
  __shared__ __align__(16) _Float16 kbuf[64 * VT_LD];   // 34 KB (K chunk / V^T chunk)
  __shared__ __align__(16) _Float16 pbuf[16 * 256];     // 8 KB  (P chunk, f16)
  __shared__ float red[16 * 8];
  __shared__ float cred[4 * 256];

  const int t = threadIdx.x, lane = t & 31, wave = t >> 5;
  const int hf = lane >> 4, n = lane & 15;
  const int bh = blockIdx.y, q0 = blockIdx.x * 16;
  const _Float16* qb  = qws + ((size_t)bh * S_ + q0) * HD_;
  const _Float16* kb  = kws + (size_t)bh * S_ * HD_;
  const _Float16* vtb = vtws + (size_t)bh * HD_ * S_;

  if (wave == 0) {
    tdm_load_1d_f16(qb, LDSOFF(qs), 16 * HD_);
    __builtin_amdgcn_s_wait_tensorcnt(0);
  }
  __syncthreads();
  const v16h aq0 = load_a_f16(qs + 0, HD_, lane);
  const v16h aq1 = load_a_f16(qs + 32, HD_, lane);

  v8f acc[16];
#pragma unroll
  for (int u = 0; u < 16; ++u) acc[u] = (v8f){};

  // ---- Phase 1: scores ----
#pragma unroll
  for (int c = 0; c < 8; ++c) {
    if (wave == 0) {
      tdm_load_1d_f16(kb + (size_t)(c * 256) * HD_, LDSOFF(kbuf), 256 * HD_);
      __builtin_amdgcn_s_wait_tensorcnt(0);
    }
    __syncthreads();
#pragma unroll
    for (int u2 = 0; u2 < 2; ++u2) {
      const int lc = wave * 32 + u2 * 16;
      v16h b0 = load_bT_f16(kbuf + (size_t)lc * HD_ + 0, HD_, lane);
      v16h b1 = load_bT_f16(kbuf + (size_t)lc * HD_ + 32, HD_, lane);
      acc[c * 2 + u2] = WMMA_F16(aq0, b0, acc[c * 2 + u2]);
      acc[c * 2 + u2] = WMMA_F16(aq1, b1, acc[c * 2 + u2]);
    }
    __syncthreads();
  }

  // ---- Phase 2: softmax (scale 1/sqrt(64) = 0.125) ----
  float rowmax[8], rowsum[8];
#pragma unroll
  for (int r = 0; r < 8; ++r) {
    float mx = -3.4e38f;
#pragma unroll
    for (int u = 0; u < 16; ++u) {
      acc[u][r] *= 0.125f;
      mx = fmaxf(mx, acc[u][r]);
    }
    mx = fmaxf(mx, __shfl_xor(mx, 1));
    mx = fmaxf(mx, __shfl_xor(mx, 2));
    mx = fmaxf(mx, __shfl_xor(mx, 4));
    mx = fmaxf(mx, __shfl_xor(mx, 8));
    rowmax[r] = mx;
  }
  if (n == 0) {
#pragma unroll
    for (int r = 0; r < 8; ++r) red[(hf * 8 + r) * 8 + wave] = rowmax[r];
  }
  __syncthreads();
#pragma unroll
  for (int r = 0; r < 8; ++r) {
    const int m = hf * 8 + r;
    float mx = red[m * 8];
#pragma unroll
    for (int w2 = 1; w2 < 8; ++w2) mx = fmaxf(mx, red[m * 8 + w2]);
    rowmax[r] = mx;
  }
  __syncthreads();
#pragma unroll
  for (int r = 0; r < 8; ++r) {
    float sm = 0.0f;
#pragma unroll
    for (int u = 0; u < 16; ++u) {
      const float e = __expf(acc[u][r] - rowmax[r]);
      acc[u][r] = e;
      sm += e;
    }
    sm += __shfl_xor(sm, 1);
    sm += __shfl_xor(sm, 2);
    sm += __shfl_xor(sm, 4);
    sm += __shfl_xor(sm, 8);
    rowsum[r] = sm;
  }
  if (n == 0) {
#pragma unroll
    for (int r = 0; r < 8; ++r) red[(hf * 8 + r) * 8 + wave] = rowsum[r];
  }
  __syncthreads();
#pragma unroll
  for (int r = 0; r < 8; ++r) {
    const int m = hf * 8 + r;
    float sm = 0.0f;
#pragma unroll
    for (int w2 = 0; w2 < 8; ++w2) sm += red[m * 8 + w2];
    rowsum[r] = 1.0f / sm;
  }
  float* arow = attn + ((size_t)bh * S_ + q0) * S_;
#pragma unroll
  for (int u = 0; u < 16; ++u) {
    const int col = (u >> 1) * 256 + wave * 32 + (u & 1) * 16 + n;
#pragma unroll
    for (int r = 0; r < 8; ++r) {
      const int m = hf * 8 + r;
      const float p = acc[u][r] * rowsum[r];
      acc[u][r] = p;
      arow[(size_t)m * S_ + col] = p;
    }
  }

  // ---- Phase 3: ctx = P @ V ----
  const int ws = wave & 3;        // head-dim col tile (16 cols)
  const int kh = wave >> 2;       // split-K half within each 256 chunk
  v8f cacc = {};
#pragma unroll
  for (int c = 0; c < 8; ++c) {
    __syncthreads();
    if (wave == 0) {
      // 2-D tile: 256 keys x 64 dims out of V^T (64 x 2048), row stride S_;
      // LDS pad: +8 dwords after every 128 dwords (row) -> VT_LD halfs/row.
      tdm_load_2d_f16(vtb + c * 256, LDSOFF(kbuf), 256, 64, S_, HD_, S_,
                      1u, 6u, 7u);
    }
#pragma unroll
    for (int u2 = 0; u2 < 2; ++u2) {
      const int lc = wave * 32 + u2 * 16 + n;
#pragma unroll
      for (int r = 0; r < 8; ++r) {
        pbuf[(hf * 8 + r) * 256 + lc] = (_Float16)acc[c * 2 + u2][r];
      }
    }
    if (wave == 0) __builtin_amdgcn_s_wait_tensorcnt(0);
    __syncthreads();
#pragma unroll
    for (int kk = 0; kk < 128; kk += 32) {
      const int kloc = kh * 128 + kk;
      v16h a  = load_a_f16(pbuf + kloc, 256, lane);
      v16h bf = load_bT_f16(kbuf + (size_t)(ws * 16) * VT_LD + kloc, VT_LD, lane);
      cacc = WMMA_F16(a, bf, cacc);
    }
  }
  __syncthreads();
  if (kh == 1) {
#pragma unroll
    for (int r = 0; r < 8; ++r) cred[ws * 256 + (hf * 8 + r) * 16 + n] = cacc[r];
  }
  __syncthreads();
  if (kh == 0) {
    const int bidx = bh >> 4, h = bh & 15;
#pragma unroll
    for (int r = 0; r < 8; ++r) {
      const int m = hf * 8 + r;
      const float v = cacc[r] + cred[ws * 256 + m * 16 + n];
      ctxws[(size_t)(bidx * S_ + q0 + m) * D_ + h * 64 + ws * 16 + n] = (_Float16)v;
    }
  }
}

// ---------------------------------------------------------------------------
// Kernel 3: out = ctx @ Wo + bo (f32). Block 128 thr: 16 rows x 64 cols.
// grid = (D/64, B*S/16)
// ---------------------------------------------------------------------------
__global__ void out_proj_kernel(const _Float16* __restrict__ ctxws,
                                const float* __restrict__ Wo, const float* __restrict__ bo,
                                float* __restrict__ out) {
  __shared__ __align__(16) _Float16 as[16 * 32];
  __shared__ __align__(16) _Float16 bsT[64 * BST_LD];

  const int t = threadIdx.x;
  const int lane = t & 31, wave = t >> 5;
  const int row0 = blockIdx.y * 16;
  const int col0 = blockIdx.x * 64 + wave * 16;
  const _Float16* Abase = ctxws + (size_t)row0 * D_;
  const float*    Bbase = Wo + blockIdx.x * 64;

  v8f acc = {};
  for (int k0 = 0; k0 < D_; k0 += 32) {
    stage_a_f16(as, Abase, D_, k0, t);
    stage_bT_f32(bsT, Bbase, D_, k0, t);
    __syncthreads();
    v16h a  = load_a_f16(as, 32, lane);
    v16h bf = load_bT_f16(bsT + (size_t)(wave * 16) * BST_LD, BST_LD, lane);
    acc = WMMA_F16(a, bf, acc);
    __syncthreads();
  }

  const int n = lane & 15;
  const float bb = bo[col0 + n];
#pragma unroll
  for (int r = 0; r < 8; ++r) {
    const int m = ((lane >> 4) << 3) + r;
    out[(size_t)(row0 + m) * D_ + (col0 + n)] = acc[r] + bb;
  }
}

// ---------------------------------------------------------------------------
extern "C" void kernel_launch(void* const* d_in, const int* in_sizes, int n_in,
                              void* d_out, int out_size, void* d_ws, size_t ws_size,
                              hipStream_t stream) {
  (void)in_sizes; (void)n_in; (void)out_size; (void)ws_size;
  const float* X  = (const float*)d_in[0];
  const float* Wq = (const float*)d_in[1];
  const float* bq = (const float*)d_in[2];
  const float* Wk = (const float*)d_in[3];
  const float* bk = (const float*)d_in[4];
  const float* Wv = (const float*)d_in[5];
  const float* bv = (const float*)d_in[6];
  const float* Wo = (const float*)d_in[7];
  const float* bo = (const float*)d_in[8];

  float* out  = (float*)d_out;
  float* attn = out + (size_t)B_ * S_ * D_;   // tuple output: [out | attn_weights]

  const size_t N = (size_t)B_ * S_ * D_;
  _Float16* qws   = (_Float16*)d_ws;
  _Float16* kws   = qws + N;
  _Float16* vtws  = kws + N;     // V stored transposed: (B,H,Hd,S)
  _Float16* ctxws = vtws + N;

  qkv_rope_kernel<<<dim3(H_, (B_ * S_) / 16, 3), 128, 0, stream>>>(
      X, Wq, bq, Wk, bk, Wv, bv, qws, kws, vtws);
  attn_fused_kernel<<<dim3(S_ / 16, B_ * H_), 256, 0, stream>>>(
      qws, kws, vtws, attn, ctxws);
  out_proj_kernel<<<dim3(D_ / 64, (B_ * S_) / 16), 128, 0, stream>>>(ctxws, Wo, bo, out);
}
